// SimpleMoEDecoderLayer_25598005084531
// MI455X (gfx1250) — matrix operand order repaired and checked by reading, not compiled, not run on hardware
//
#include <hip/hip_runtime.h>
#include <hip/hip_bf16.h>

// ---------------- problem constants ----------------
#define BB 4
#define SS 1024
#define DD 1024
#define HH 16
#define HDIM 64
#define FF 4096
#define EE 8
#define TT (BB*SS)      // 4096 tokens
#define BHH (BB*HH)     // 64 (b,h) pairs

typedef _Float16 h8   __attribute__((ext_vector_type(8)));
typedef _Float16 v16h __attribute__((ext_vector_type(16)));
typedef float    v8f  __attribute__((ext_vector_type(8)));

#define BM  64
#define BKK 32
#define ASTR 40   // LDS halfs per A row (pad, keeps 16B alignment)

// CDNA5 async copy: global -> LDS, tracked by ASYNCcnt (inline asm per bridge doc)
__device__ __forceinline__ void async_b128(const void* g, const void* lds) {
  unsigned loff = (unsigned)(size_t)lds;   // low 32 bits of generic shared addr = LDS offset
  asm volatile("global_load_async_to_lds_b128 %0, %1, off" :: "v"(loff), "v"(g) : "memory");
}
__device__ __forceinline__ void wait_async0() {
  asm volatile("s_wait_asynccnt 0x0" ::: "memory");
}

// ---------------- generic WMMA GEMM (B pre-transposed: Bt is [K][N]) ----------------
// C[m,n] = alpha * sum_k A[m,k]*Bt[k,n] + bias[n]   (optional ReLU)
// f16 A/Bt, f32 accum; out to f32 C and/or f16 Ch. Batch via blockIdx.z strides.
// MoE mode: cnt_ptr/off_ptr give dynamic M per z; row_idx gathers A rows;
// C rows packed at off+m. Double-buffered LDS fed by async-to-LDS copies.
template<int BN_, int WN_>
__launch_bounds__(256)
__global__ void gemm_wmma_f16(
    const _Float16* __restrict__ A, long lda, long strideAb,
    const _Float16* __restrict__ Bt, long ldbt, long strideBb,
    const float* __restrict__ bias, long strideBias,
    float* __restrict__ C, _Float16* __restrict__ Ch, long ldc, long strideCb,
    int M, int N, int K, float alpha, int do_relu,
    const int* __restrict__ row_idx, const int* __restrict__ cnt_ptr,
    const int* __restrict__ off_ptr)
{
  constexpr int WM_   = 8 / WN_;        // waves along M
  constexpr int WAVEM = BM / WM_;       // rows per wave
  constexpr int WAVEN = BN_ / WN_;      // cols per wave
  constexpr int MI    = WAVEM / 16;
  constexpr int NI    = WAVEN / 16;
  constexpr int BSTR  = BN_ + 8;        // LDS halfs per Bt row (pad; multiple of 8)

  __shared__ _Float16 Als[2][BM * ASTR];
  __shared__ _Float16 Bls[2][BKK * BSTR];

  const int z  = blockIdx.z;
  const int m0 = blockIdx.y * BM;
  const int n0 = blockIdx.x * BN_;

  int Meff = M, rb = 0;
  if (cnt_ptr) { Meff = cnt_ptr[z]; rb = off_ptr[z]; if (m0 >= Meff) return; }

  const _Float16* Ab = A  + (cnt_ptr ? 0 : (long)z * strideAb);
  const _Float16* Bb = Bt + (long)z * strideBb;

  const int t    = threadIdx.x;
  const int lane = t & 31;
  const int wave = t >> 5;
  const int wm   = wave % WM_;
  const int wn   = wave / WM_;
  const int half = lane >> 4;
  const int l15  = lane & 15;

  // A stager: 64 rows x 32 halfs, one 16B chunk / thread
  const int ar  = t >> 2;
  const int acb = (t & 3) * 8;
  // Bt stager: 32 rows x BN_ halfs, BN_/64 chunks / thread
  constexpr int BCH = BN_ / 8;          // 16B chunks per Bt row

  const _Float16* aptr = nullptr;
  {
    int gm = m0 + ar;
    if (gm < Meff) {
      long arow = cnt_ptr ? (long)(row_idx ? row_idx[rb + gm] : (rb + gm))
                          : (long)gm;
      aptr = Ab + arow * lda + acb;
    }
  }

  auto issue = [&](int k0, int buf) {
    if (aptr) {
      async_b128(aptr + k0, &Als[buf][ar * ASTR + acb]);
    } else {
      h8 zz = {};
      *(h8*)(&Als[buf][ar * ASTR + acb]) = zz;   // zero-pad invalid rows
    }
    #pragma unroll
    for (int c = 0; c < BN_ / 64; ++c) {
      int u   = t + c * 256;
      int bk  = u / BCH;
      int bn8 = (u % BCH) * 8;
      async_b128(Bb + (long)(k0 + bk) * ldbt + n0 + bn8,
                 &Bls[buf][bk * BSTR + bn8]);
    }
  };

  v8f acc[MI][NI];
  { v8f zz = {}; 
    #pragma unroll
    for (int mi = 0; mi < MI; ++mi)
      #pragma unroll
      for (int ni = 0; ni < NI; ++ni) acc[mi][ni] = zz; }

  issue(0, 0);
  const int nk = K / BKK;
  for (int i = 0; i < nk; ++i) {
    wait_async0();          // tile i landed in LDS
    __syncthreads();        // visible to all waves; prev buffer free
    if (i + 1 < nk) issue((i + 1) * BKK, (i + 1) & 1);
    const int cur = i & 1;

    v16h af[MI];
    #pragma unroll
    for (int mi = 0; mi < MI; ++mi) {
      const _Float16* ap = &Als[cur][(wm * WAVEM + mi * 16 + l15) * ASTR + 8 * half];
      h8 lo = *(const h8*)ap;          // K = 8*half .. +7
      h8 hi = *(const h8*)(ap + 16);   // K = 16+8*half .. +7
      #pragma unroll
      for (int e = 0; e < 8; ++e) { af[mi][e] = lo[e]; af[mi][8 + e] = hi[e]; }
    }
    #pragma unroll
    for (int ni = 0; ni < NI; ++ni) {
      const _Float16* bp = &Bls[cur][(8 * half) * BSTR + wn * WAVEN + ni * 16 + l15];
      v16h bf;
      #pragma unroll
      for (int e = 0; e < 8; ++e) { bf[e] = bp[e * BSTR]; bf[8 + e] = bp[(16 + e) * BSTR]; }
      #pragma unroll
      for (int mi = 0; mi < MI; ++mi)
        acc[mi][ni] = __builtin_amdgcn_wmma_f32_16x16x32_f16(
            false, af[mi], false, bf, (short)0, acc[mi][ni], false, false);
    }
  }

  // ---- epilogue: C/D layout (lane n=l15, VGPR r -> m = r + 8*half) ----
  const long cz = cnt_ptr ? 0 : (long)z * strideCb;
  float*     Cb  = C  ? C  + cz : nullptr;
  _Float16*  Chb = Ch ? Ch + cz : nullptr;
  const float* bb = bias ? bias + (long)z * strideBias : nullptr;

  #pragma unroll
  for (int mi = 0; mi < MI; ++mi) {
    #pragma unroll
    for (int r = 0; r < 8; ++r) {
      int gm = m0 + wm * WAVEM + mi * 16 + r + 8 * half;
      if (gm >= Meff) continue;
      long crow = cnt_ptr ? (long)(rb + gm) : (long)gm;
      #pragma unroll
      for (int ni = 0; ni < NI; ++ni) {
        int n = n0 + wn * WAVEN + ni * 16 + l15;
        float v = acc[mi][ni][r] * alpha + (bb ? bb[n] : 0.f);
        if (do_relu && v < 0.f) v = 0.f;
        if (Cb)  Cb[crow * ldc + n]  = v;
        if (Chb) Chb[crow * ldc + n] = (_Float16)v;
      }
    }
  }
}

// ---------------- f32 -> f16 convert ----------------
__global__ void cvt_f16(const float* __restrict__ s, _Float16* __restrict__ d, long n) {
  long i = (long)blockIdx.x * blockDim.x + threadIdx.x;
  long st = (long)gridDim.x * blockDim.x;
  for (; i < n; i += st) d[i] = (_Float16)s[i];
}

// ---------------- f32 -> f16 transpose: dst[k*N+n] = src[n*K+k], batched ----------------
__global__ void cvt_transpose(const float* __restrict__ src, _Float16* __restrict__ dst,
                              int Nn, int Kk, long sS, long sD) {
  __shared__ float tile[32][33];
  int z = blockIdx.z;
  int n0 = blockIdx.x * 32, k0 = blockIdx.y * 32;
  int tx = threadIdx.x & 31, ty = threadIdx.x >> 5;   // 32x8
  const float* S = src + (long)z * sS;
  _Float16*    Dp = dst + (long)z * sD;
  for (int i = ty; i < 32; i += 8) tile[i][tx] = S[(long)(n0 + i) * Kk + k0 + tx];
  __syncthreads();
  for (int i = ty; i < 32; i += 8) Dp[(long)(k0 + i) * Nn + n0 + tx] = (_Float16)tile[tx][i];
}

// ---------------- qkv split: Q [bh][s][hd], K^T [bh][hd][s], V [bh][s][hd] ----------------
__global__ void split_qkv(const _Float16* __restrict__ qkv,
                          _Float16* __restrict__ q, _Float16* __restrict__ kt,
                          _Float16* __restrict__ v) {
  int tkn = blockIdx.x; int b = tkn >> 10; int s = tkn & (SS - 1);
  const _Float16* src = qkv + (long)tkn * (3 * DD);
  for (int d = threadIdx.x; d < DD; d += 256) {
    int h = d >> 6, hd = d & 63;
    long bh = (long)(b * HH + h);
    q[(bh * SS + s) * HDIM + hd]  = src[d];
    kt[(bh * HDIM + hd) * SS + s] = src[DD + d];
    v[(bh * SS + s) * HDIM + hd]  = src[2 * DD + d];
  }
}

// ---------------- masked softmax, in place on f16 scores ----------------
__global__ void softmax_mask(_Float16* __restrict__ sc, const int* __restrict__ mask) {
  int q = blockIdx.x, bh = blockIdx.y, b = bh >> 4;
  _Float16* row = sc + ((long)bh * SS + q) * SS;
  const int* mrow = mask + b * SS;
  __shared__ float red[256];
  int t = threadIdx.x;
  float vals[4], mx = -1e30f;
  #pragma unroll
  for (int i = 0; i < 4; ++i) {
    int kx = t + i * 256;
    float v = (float)row[kx];
    if (mrow[kx] == 0) v = -1e30f;
    vals[i] = v; mx = fmaxf(mx, v);
  }
  red[t] = mx; __syncthreads();
  for (int s = 128; s > 0; s >>= 1) { if (t < s) red[t] = fmaxf(red[t], red[t + s]); __syncthreads(); }
  mx = red[0]; __syncthreads();
  float sum = 0.f;
  #pragma unroll
  for (int i = 0; i < 4; ++i) { float e = __expf(vals[i] - mx); vals[i] = e; sum += e; }
  red[t] = sum; __syncthreads();
  for (int s = 128; s > 0; s >>= 1) { if (t < s) red[t] += red[t + s]; __syncthreads(); }
  float inv = 1.f / red[0];
  #pragma unroll
  for (int i = 0; i < 4; ++i) { int kx = t + i * 256; row[kx] = (_Float16)(vals[i] * inv); }
}

// ---------------- merge heads -> f16 (B*S, D) ----------------
__global__ void merge_heads(const float* __restrict__ ao, _Float16* __restrict__ out) {
  int tkn = blockIdx.x; int b = tkn >> 10; int s = tkn & (SS - 1);
  for (int d = threadIdx.x; d < DD; d += 256) {
    int h = d >> 6, hd = d & 63;
    out[(long)tkn * DD + d] =
        (_Float16)ao[(((long)(b * HH + h)) * SS + s) * HDIM + hd];
  }
}

// ---------------- residual add + layernorm (f32 out + optional f16) ----------------
__global__ void add_ln(const float* __restrict__ xa, const float* __restrict__ xb,
                       const float* __restrict__ g, const float* __restrict__ bt,
                       float* __restrict__ y, _Float16* __restrict__ yh) {
  int tkn = blockIdx.x, t = threadIdx.x;
  const float* A = xa + (long)tkn * DD;
  const float* Bv = xb + (long)tkn * DD;
  __shared__ float red[256];
  float v[4], s = 0.f;
  #pragma unroll
  for (int i = 0; i < 4; ++i) { v[i] = A[t + i * 256] + Bv[t + i * 256]; s += v[i]; }
  red[t] = s; __syncthreads();
  for (int k = 128; k > 0; k >>= 1) { if (t < k) red[t] += red[t + k]; __syncthreads(); }
  float mean = red[0] * (1.f / DD); __syncthreads();
  float s2 = 0.f;
  #pragma unroll
  for (int i = 0; i < 4; ++i) { float d = v[i] - mean; s2 += d * d; }
  red[t] = s2; __syncthreads();
  for (int k = 128; k > 0; k >>= 1) { if (t < k) red[t] += red[t + k]; __syncthreads(); }
  float rstd = rsqrtf(red[0] * (1.f / DD) + 1e-5f);
  #pragma unroll
  for (int i = 0; i < 4; ++i) {
    int d = t + i * 256;
    float o = (v[i] - mean) * rstd * g[d] + bt[d];
    y[(long)tkn * DD + d] = o;
    if (yh) yh[(long)tkn * DD + d] = (_Float16)o;
  }
}

// ---------------- gating: logits, softmax(8), top-2, counts ----------------
__launch_bounds__(128)
__global__ void gate_topk(const float* __restrict__ x1, const float* __restrict__ gw,
                          const float* __restrict__ gb, int* __restrict__ counts,
                          int* __restrict__ tok_e, float* __restrict__ tok_w) {
  int tkn = blockIdx.x, t = threadIdx.x;
  __shared__ float red[EE][128];
  float acc[EE] = {};
  const float* xr = x1 + (long)tkn * DD;
  for (int d = t; d < DD; d += 128) {
    float xv = xr[d];
    #pragma unroll
    for (int e = 0; e < EE; ++e) acc[e] += xv * gw[e * DD + d];
  }
  #pragma unroll
  for (int e = 0; e < EE; ++e) red[e][t] = acc[e];
  __syncthreads();
  for (int k = 64; k > 0; k >>= 1) {
    if (t < k) {
      #pragma unroll
      for (int e = 0; e < EE; ++e) red[e][t] += red[e][t + k];
    }
    __syncthreads();
  }
  if (t == 0) {
    float lg[EE], mx = -1e30f;
    #pragma unroll
    for (int e = 0; e < EE; ++e) { lg[e] = red[e][0] + gb[e]; mx = fmaxf(mx, lg[e]); }
    float sum = 0.f;
    #pragma unroll
    for (int e = 0; e < EE; ++e) { lg[e] = __expf(lg[e] - mx); sum += lg[e]; }
    #pragma unroll
    for (int e = 0; e < EE; ++e) lg[e] /= sum;
    int e0 = 0;
    for (int e = 1; e < EE; ++e) if (lg[e] > lg[e0]) e0 = e;
    int e1 = (e0 == 0) ? 1 : 0;
    for (int e = 0; e < EE; ++e) { if (e == e0) continue; if (lg[e] > lg[e1]) e1 = e; }
    float v0 = lg[e0], v1 = lg[e1], ws = v0 + v1 + 1e-9f;
    tok_e[2 * tkn] = e0; tok_e[2 * tkn + 1] = e1;
    tok_w[2 * tkn] = v0 / ws; tok_w[2 * tkn + 1] = v1 / ws;
    atomicAdd(&counts[e0], 1); atomicAdd(&counts[e1], 1);
  }
}

__global__ void moe_init(int* counts) { if (threadIdx.x < EE) counts[threadIdx.x] = 0; }

__global__ void moe_scan(const int* __restrict__ counts, int* __restrict__ offsets,
                         int* __restrict__ cursors) {
  if (threadIdx.x == 0 && blockIdx.x == 0) {
    int acc = 0;
    for (int e = 0; e < EE; ++e) { offsets[e] = acc; cursors[e] = acc; acc += counts[e]; }
  }
}

__global__ void moe_assign(const int* __restrict__ tok_e, int* __restrict__ cursors,
                           int* __restrict__ expert_rows, int* __restrict__ token_slot) {
  int tkn = blockIdx.x * blockDim.x + threadIdx.x;
  if (tkn >= TT) return;
  for (int j = 0; j < 2; ++j) {
    int e = tok_e[2 * tkn + j];
    int row = atomicAdd(&cursors[e], 1);
    expert_rows[row] = tkn;
    token_slot[2 * tkn + j] = row;
  }
}

// ---------------- top-2 weighted combine + residual + LN2 -> output ----------------
__global__ void combine_ln2(const float* __restrict__ x1, const float* __restrict__ ob,
                            const int* __restrict__ token_slot, const float* __restrict__ tok_w,
                            const float* __restrict__ g, const float* __restrict__ bt,
                            float* __restrict__ y) {
  int tkn = blockIdx.x, t = threadIdx.x;
  long s0 = token_slot[2 * tkn], s1 = token_slot[2 * tkn + 1];
  float w0 = tok_w[2 * tkn], w1 = tok_w[2 * tkn + 1];
  __shared__ float red[256];
  float v[4], s = 0.f;
  #pragma unroll
  for (int i = 0; i < 4; ++i) {
    int d = t + i * 256;
    v[i] = x1[(long)tkn * DD + d] + w0 * ob[s0 * DD + d] + w1 * ob[s1 * DD + d];
    s += v[i];
  }
  red[t] = s; __syncthreads();
  for (int k = 128; k > 0; k >>= 1) { if (t < k) red[t] += red[t + k]; __syncthreads(); }
  float mean = red[0] * (1.f / DD); __syncthreads();
  float s2 = 0.f;
  #pragma unroll
  for (int i = 0; i < 4; ++i) { float d = v[i] - mean; s2 += d * d; }
  red[t] = s2; __syncthreads();
  for (int k = 128; k > 0; k >>= 1) { if (t < k) red[t] += red[t + k]; __syncthreads(); }
  float rstd = rsqrtf(red[0] * (1.f / DD) + 1e-5f);
  #pragma unroll
  for (int i = 0; i < 4; ++i) {
    int d = t + i * 256;
    y[(long)tkn * DD + d] = (v[i] - mean) * rstd * g[d] + bt[d];
  }
}

// ---------------- host side ----------------
extern "C" void kernel_launch(void* const* d_in, const int* in_sizes, int n_in,
                              void* d_out, int out_size, void* d_ws, size_t ws_size,
                              hipStream_t stream) {
  const float* x        = (const float*)d_in[0];
  const int*   mask     = (const int*)d_in[1];
  const float* in_w     = (const float*)d_in[2];
  const float* in_b     = (const float*)d_in[3];
  const float* out_w    = (const float*)d_in[4];
  const float* out_b    = (const float*)d_in[5];
  const float* ln1_g    = (const float*)d_in[6];
  const float* ln1_b    = (const float*)d_in[7];
  const float* ln2_g    = (const float*)d_in[8];
  const float* ln2_b    = (const float*)d_in[9];
  const float* gate_w   = (const float*)d_in[10];
  const float* gate_b   = (const float*)d_in[11];
  const float* w1       = (const float*)d_in[12];
  const float* b1       = (const float*)d_in[13];
  const float* w2       = (const float*)d_in[14];
  const float* b2       = (const float*)d_in[15];
  float* out = (float*)d_out;

  char* p = (char*)d_ws;
  auto alloc = [&](size_t bytes) -> void* {
    void* r = (void*)p; p += (bytes + 255) & ~(size_t)255; return r;
  };
  // all weight copies transposed to [K][N] f16
  _Float16* w_in_t  = (_Float16*)alloc((size_t)DD * 3 * DD * 2);
  _Float16* w_out_t = (_Float16*)alloc((size_t)DD * DD * 2);
  _Float16* w1_t    = (_Float16*)alloc((size_t)EE * DD * FF * 2);
  _Float16* w2_t    = (_Float16*)alloc((size_t)EE * FF * DD * 2);
  _Float16* x_h     = (_Float16*)alloc((size_t)TT * DD * 2);
  _Float16* qkv_h   = (_Float16*)alloc((size_t)TT * 3 * DD * 2);
  _Float16* q_h     = (_Float16*)alloc((size_t)BHH * SS * HDIM * 2);
  _Float16* k_t     = (_Float16*)alloc((size_t)BHH * HDIM * SS * 2);
  _Float16* v_h     = (_Float16*)alloc((size_t)BHH * SS * HDIM * 2);
  _Float16* sc_h    = (_Float16*)alloc((size_t)BHH * SS * SS * 2);   // 128 MB
  float*    attn_o  = (float*)alloc((size_t)BHH * SS * HDIM * 4);
  _Float16* attn_h  = (_Float16*)alloc((size_t)TT * DD * 2);
  float*    attn_p  = (float*)alloc((size_t)TT * DD * 4);
  float*    x1      = (float*)alloc((size_t)TT * DD * 4);
  _Float16* x1_h    = (_Float16*)alloc((size_t)TT * DD * 2);
  _Float16* hbuf    = (_Float16*)alloc((size_t)2 * TT * FF * 2);     // 64 MB
  float*    obuf    = (float*)alloc((size_t)2 * TT * DD * 4);        // 32 MB
  int* counts      = (int*)alloc(EE * 4);
  int* offsets     = (int*)alloc(EE * 4);
  int* cursors     = (int*)alloc(EE * 4);
  int* expert_rows = (int*)alloc(2 * TT * 4);
  int* tok_e       = (int*)alloc(2 * TT * 4);
  float* tok_w     = (float*)alloc(2 * TT * 4);
  int* token_slot  = (int*)alloc(2 * TT * 4);
  (void)ws_size; (void)in_sizes; (void)n_in; (void)out_size;

  // 1) convert + transpose weights to f16 [K][N]; convert x to f16
  cvt_transpose<<<dim3(3 * DD / 32, DD / 32, 1), 256, 0, stream>>>(
      in_w, w_in_t, 3 * DD, DD, 0, 0);
  cvt_transpose<<<dim3(DD / 32, DD / 32, 1), 256, 0, stream>>>(
      out_w, w_out_t, DD, DD, 0, 0);
  cvt_transpose<<<dim3(FF / 32, DD / 32, EE), 256, 0, stream>>>(
      w1, w1_t, FF, DD, (long)FF * DD, (long)FF * DD);
  cvt_transpose<<<dim3(DD / 32, FF / 32, EE), 256, 0, stream>>>(
      w2, w2_t, DD, FF, (long)DD * FF, (long)DD * FF);
  cvt_f16<<<4096, 256, 0, stream>>>(x, x_h, (long)TT * DD);

  // 2) QKV projection: (T,3D) = x_h (T,D) @ in_w^T, +bias, f16 out
  gemm_wmma_f16<128, 4><<<dim3(3 * DD / 128, TT / BM, 1), 256, 0, stream>>>(
      x_h, DD, 0, w_in_t, 3 * DD, 0, in_b, 0,
      nullptr, qkv_h, 3 * DD, 0, TT, 3 * DD, DD, 1.f, 0,
      nullptr, nullptr, nullptr);

  // 3) split into per-head Q, K^T, V
  split_qkv<<<TT, 256, 0, stream>>>(qkv_h, q_h, k_t, v_h);

  // 4) scores = Q @ K^T / sqrt(HD), batched over 64 (b,h), f16 out
  gemm_wmma_f16<128, 4><<<dim3(SS / 128, SS / BM, BHH), 256, 0, stream>>>(
      q_h, HDIM, (long)SS * HDIM, k_t, SS, (long)HDIM * SS, nullptr, 0,
      nullptr, sc_h, SS, (long)SS * SS, SS, SS, HDIM, 0.125f, 0,
      nullptr, nullptr, nullptr);

  // 5) masked softmax in place
  softmax_mask<<<dim3(SS, BHH), 256, 0, stream>>>(sc_h, mask);

  // 6) attn = P @ V, batched (Bt = V natural [s][hd] layout), N=64 variant
  gemm_wmma_f16<64, 2><<<dim3(1, SS / BM, BHH), 256, 0, stream>>>(
      sc_h, SS, (long)SS * SS, v_h, HDIM, (long)SS * HDIM, nullptr, 0,
      attn_o, nullptr, HDIM, (long)SS * HDIM, SS, HDIM, SS, 1.f, 0,
      nullptr, nullptr, nullptr);

  // 7) merge heads -> f16, out projection
  merge_heads<<<TT, 256, 0, stream>>>(attn_o, attn_h);
  gemm_wmma_f16<128, 4><<<dim3(DD / 128, TT / BM, 1), 256, 0, stream>>>(
      attn_h, DD, 0, w_out_t, DD, 0, out_b, 0,
      attn_p, nullptr, DD, 0, TT, DD, DD, 1.f, 0,
      nullptr, nullptr, nullptr);

  // 8) x1 = LN1(x + attn_proj) (f32 + f16 copy)
  add_ln<<<TT, 256, 0, stream>>>(x, attn_p, ln1_g, ln1_b, x1, x1_h);

  // 9) gating + token->expert grouping
  moe_init<<<1, 32, 0, stream>>>(counts);
  gate_topk<<<TT, 128, 0, stream>>>(x1, gate_w, gate_b, counts, tok_e, tok_w);
  moe_scan<<<1, 32, 0, stream>>>(counts, offsets, cursors);
  moe_assign<<<(TT + 255) / 256, 256, 0, stream>>>(tok_e, cursors, expert_rows, token_slot);

  // 10) MoE GEMM1: h = relu(gathered x1 @ w1[e]^T + b1[e]), f16 out
  gemm_wmma_f16<128, 4><<<dim3(FF / 128, TT / BM, EE), 256, 0, stream>>>(
      x1_h, DD, 0, w1_t, FF, (long)DD * FF, b1, FF,
      nullptr, hbuf, FF, 0, TT, FF, DD, 1.f, 1,
      expert_rows, counts, offsets);

  // 11) MoE GEMM2: out = h @ w2[e]^T + b2[e], f32 out (packed rows)
  gemm_wmma_f16<128, 4><<<dim3(DD / 128, TT / BM, EE), 256, 0, stream>>>(
      hbuf, FF, 0, w2_t, DD, (long)FF * DD, b2, DD,
      obuf, nullptr, DD, 0, TT, DD, FF, 1.f, 0,
      nullptr, counts, offsets);

  // 12) top-2 weighted combine + residual + LN2 -> final output
  combine_ln2<<<TT, 256, 0, stream>>>(x1, obuf, token_slot, tok_w, ln2_g, ln2_b, out);
}
